// AnomalyAttention_36653250904476
// MI455X (gfx1250) — matrix operand order repaired and checked by reading, not compiled, not run on hardware
//
#include <hip/hip_runtime.h>
#include <hip/hip_bf16.h>

#define BB   4
#define TT   1024
#define DD   512
#define HH   8
#define DHH  64

typedef __attribute__((ext_vector_type(16))) _Float16 v16h;
typedef __attribute__((ext_vector_type(8)))  float    v8f;
typedef __attribute__((ext_vector_type(4)))  int      i4v;

#if defined(__has_builtin)
#if __has_builtin(__builtin_amdgcn_global_load_async_to_lds_b128) && \
    __has_builtin(__builtin_amdgcn_s_wait_asynccnt)
#define HAVE_ASYNC_LDS 1
#endif
#endif
#ifndef HAVE_ASYNC_LDS
#define HAVE_ASYNC_LDS 0
#endif

// ---------------------------------------------------------------------------
// WMMA fragment helpers (wave32, per CDNA5 ISA 7.12.2 layouts)
// ---------------------------------------------------------------------------
__device__ __forceinline__ v8f wmma_f16(v16h a, v16h b, v8f c) {
    // 8 args: (neg_a, A, neg_b, B, c_mod, C, reuse_a, reuse_b)
    return __builtin_amdgcn_wmma_f32_16x16x32_f16(false, a, false, b,
                                                  (short)0, c, false, false);
}

// A (16x32 f16), row-major source, lda in elements.
__device__ __forceinline__ v16h load_a_f16(const _Float16* __restrict__ A,
                                           int lda, int m0, int k0, int lane) {
    union { v16h v; unsigned int u[8]; } f;
    const int lo = (lane < 16) ? 0 : 8;
    const _Float16* row = A + (size_t)(m0 + (lane & 15)) * lda + k0;
#pragma unroll
    for (int j = 0; j < 8; ++j) {
        const int kb = 2 * j + ((j >= 4) ? 8 : 0) + lo;
        f.u[j] = *(const unsigned int*)(row + kb);
    }
    return f.v;
}

// A fragment converted on the fly from f32 source (for series @ V).
__device__ __forceinline__ v16h load_a_f32(const float* __restrict__ A,
                                           int lda, int m0, int k0, int lane) {
    v16h v;
    const int lo = (lane < 16) ? 0 : 8;
    const float* row = A + (size_t)(m0 + (lane & 15)) * lda + k0;
#pragma unroll
    for (int j = 0; j < 8; ++j) {
        const int kb = 2 * j + ((j >= 4) ? 8 : 0) + lo;
        float2 p = *(const float2*)(row + kb);
        v[2 * j]     = (_Float16)p.x;
        v[2 * j + 1] = (_Float16)p.y;
    }
    return v;
}

// B (32x16 f16) from TRANSPOSED storage Bt[N][K] (works for global or LDS).
__device__ __forceinline__ v16h load_b_f16t(const _Float16* Bt,
                                            int ldb, int n0, int k0, int lane) {
    union { v16h v; unsigned int u[8]; } f;
    const int lo = (lane < 16) ? 0 : 8;
    const _Float16* row = Bt + (size_t)(n0 + (lane & 15)) * ldb + k0 + lo;
#pragma unroll
    for (int j = 0; j < 8; ++j) f.u[j] = *(const unsigned int*)(row + 2 * j + ((j >= 4) ? 8 : 0));
    return f.v;
}

// 16-byte global -> LDS staging (async on gfx1250 when available).
__device__ __forceinline__ void stage16(const _Float16* src, _Float16* dst) {
#if HAVE_ASYNC_LDS
    __builtin_amdgcn_global_load_async_to_lds_b128(
        (__attribute__((address_space(1))) i4v*)(void*)src,
        (__attribute__((address_space(3))) i4v*)(void*)dst, 0, 0);
#else
    *(float4*)dst = *(const float4*)src;
#endif
}
__device__ __forceinline__ void stage_wait() {
#if HAVE_ASYNC_LDS
    __builtin_amdgcn_s_wait_asynccnt(0);
#endif
    __syncthreads();
}

__device__ __forceinline__ float wave_max(float v) {
#pragma unroll
    for (int off = 16; off > 0; off >>= 1) v = fmaxf(v, __shfl_xor(v, off, 32));
    return v;
}
__device__ __forceinline__ float wave_sum(float v) {
#pragma unroll
    for (int off = 16; off > 0; off >>= 1) v += __shfl_xor(v, off, 32);
    return v;
}

// ---------------------------------------------------------------------------
// Precision / layout transforms
// ---------------------------------------------------------------------------
__global__ void k_f32_to_f16(const float* __restrict__ in,
                             _Float16* __restrict__ out, int n) {
    int i = blockIdx.x * 256 + threadIdx.x;
    if (i < n) out[i] = (_Float16)in[i];
}

// Wt[n*512 + k] = W[k*512 + n]  (f32 -> f16, transpose so B frags read rows)
__global__ void k_transpose_w(const float* __restrict__ W,
                              _Float16* __restrict__ Wt) {
    int i = blockIdx.x * 256 + threadIdx.x;   // 512*512 elements
    int k = i & 511;
    int n = i >> 9;
    Wt[i] = (_Float16)W[k * DD + n];
}

// ---------------------------------------------------------------------------
// WMMA GEMM: Y[m,n] = sum_k Xh[m,k]*Wt[n,k] + bias[n]
// Block = 8 waves = 8 M-tiles (M=128) sharing ONE 64-wide N strip.
// The B strip is staged through LDS (double-buffered, async when available),
// so weights are read from L2 once per block instead of once per wave.
// mode 0: f16 out, head-major [B,H,T,DH] (Q,K); mode 1: [B,H,DH,T] (V);
// mode 3: f32 out, plain [M,N] (final projection)
// ---------------------------------------------------------------------------
__global__ void k_gemm(const _Float16* __restrict__ Xh,
                       const _Float16* __restrict__ Wt,
                       const float* __restrict__ bias,
                       _Float16* __restrict__ out_h,
                       float* __restrict__ out_f,
                       int M, int N, int K, int mode) {
    __shared__ _Float16 lb[2][64 * 32];           // 2 x 4KB B chunks
    const int tid  = threadIdx.x;
    const int lane = tid & 31;
    const int w    = tid >> 5;
    const int strips_n = N >> 6;
    const int n0 = (blockIdx.x % strips_n) << 6;
    const int m0 = ((blockIdx.x / strips_n) << 7) + (w << 4);

    // this thread's staging slot: row (of 64) x 8-element column group
    const int srow = tid >> 2, scol = (tid & 3) << 3;
    const _Float16* bsrc = Wt + (size_t)(n0 + srow) * K + scol;
    _Float16* bdst0 = &lb[0][srow * 32 + scol];
    _Float16* bdst1 = &lb[1][srow * 32 + scol];

    const int nch = K >> 5;                        // 32-wide K chunks
    stage16(bsrc, bdst0);
    stage_wait();

    v8f acc[4] = {};
    for (int c = 0; c < nch; ++c) {
        const int cur = c & 1;
        if (c + 1 < nch) stage16(bsrc + (c + 1) * 32, cur ? bdst0 : bdst1);
        const v16h a = load_a_f16(Xh, K, m0, c * 32, lane);
#pragma unroll
        for (int s = 0; s < 4; ++s) {
            const v16h b = load_b_f16t(&lb[cur][0], 32, 16 * s, 0, lane);
            acc[s] = wmma_f16(a, b, acc[s]);
        }
        stage_wait();
    }

    const int lo = (lane < 16) ? 0 : 8;
#pragma unroll
    for (int s = 0; s < 4; ++s) {
        const int n  = n0 + 16 * s + (lane & 15);
        const float bv = bias[n];
#pragma unroll
        for (int r = 0; r < 8; ++r) {
            const int m = m0 + r + lo;
            const float val = acc[s][r] + bv;
            if (mode == 0) {
                const int b_ = m >> 10, t = m & 1023, h = n >> 6, dh = n & 63;
                out_h[(((size_t)(b_ * HH + h) * TT + t) * DHH) + dh] = (_Float16)val;
            } else if (mode == 1) {
                const int b_ = m >> 10, t = m & 1023, h = n >> 6, dh = n & 63;
                out_h[(((size_t)(b_ * HH + h) * DHH + dh) * TT) + t] = (_Float16)val;
            } else {
                out_f[(size_t)m * N + n] = val;
            }
        }
    }
}

// ---------------------------------------------------------------------------
// logits[b,h,t,s] = (q . k) / sqrt(DH) -> series region.
// Block = 8 waves = 8 t-tiles (128 rows) sharing one 64-wide s strip.
// The K-strip (64 x 64 f16 = 8KB) is staged into LDS ONCE per block.
// ---------------------------------------------------------------------------
__global__ void k_qk_logits(const _Float16* __restrict__ qh,
                            const _Float16* __restrict__ kh,
                            float* __restrict__ series) {
    __shared__ _Float16 lb[64 * 64];               // full K strip
    const int bh   = blockIdx.z;
    const int tid  = threadIdx.x;
    const int lane = tid & 31;
    const int w    = tid >> 5;
    const int m0 = ((blockIdx.x >> 4) << 7) + (w << 4);   // 8 m-groups of 128
    const int n0 = (blockIdx.x & 15) << 6;                // 16 s strips
    const _Float16* qb = qh + (size_t)bh * TT * DHH;
    const _Float16* kb = kh + (size_t)bh * TT * DHH;

    const int srow = tid >> 2, scol = (tid & 3) << 3;
#pragma unroll
    for (int cc = 0; cc < 2; ++cc)
        stage16(kb + (size_t)(n0 + srow) * DHH + scol + 32 * cc,
                &lb[srow * 64 + scol + 32 * cc]);
    stage_wait();

    v8f acc[4] = {};
#pragma unroll
    for (int k0 = 0; k0 < DHH; k0 += 32) {
        const v16h a = load_a_f16(qb, DHH, m0, k0, lane);
#pragma unroll
        for (int s = 0; s < 4; ++s) {
            const v16h b = load_b_f16t(lb, DHH, 16 * s, k0, lane);
            acc[s] = wmma_f16(a, b, acc[s]);
        }
    }

    float* srow_p = series + (size_t)bh * TT * TT;
    const int lo = (lane < 16) ? 0 : 8;
#pragma unroll
    for (int s = 0; s < 4; ++s) {
        const int n = n0 + 16 * s + (lane & 15);
#pragma unroll
        for (int r = 0; r < 8; ++r)
            srow_p[(size_t)(m0 + r + lo) * TT + n] = acc[s][r] * 0.125f;
    }
}

// ---------------------------------------------------------------------------
// In-place row softmax over the last axis (row length 1024)
// ---------------------------------------------------------------------------
__global__ void k_softmax(float* __restrict__ series) {
    __shared__ float buf[TT];
    __shared__ float red[8];
    float* p = series + (size_t)blockIdx.x * TT;
    const int tid = threadIdx.x;

    float m = -3.0e38f;
    for (int i = tid; i < TT; i += 256) { float v = p[i]; buf[i] = v; m = fmaxf(m, v); }
    m = wave_max(m);
    if ((tid & 31) == 0) red[tid >> 5] = m;
    __syncthreads();
    float rm = red[0];
#pragma unroll
    for (int i = 1; i < 8; ++i) rm = fmaxf(rm, red[i]);
    __syncthreads();

    float s = 0.f;
    for (int i = tid; i < TT; i += 256) { float e = __expf(buf[i] - rm); buf[i] = e; s += e; }
    s = wave_sum(s);
    if ((tid & 31) == 0) red[tid >> 5] = s;
    __syncthreads();
    float tot = red[0];
#pragma unroll
    for (int i = 1; i < 8; ++i) tot += red[i];
    const float inv = 1.0f / tot;
    for (int i = tid; i < TT; i += 256) p[i] = buf[i] * inv;
}

// ---------------------------------------------------------------------------
// sigma[b,h,t] = softplus(x @ Ws + b)[b,t,h] + 1e-6
// ---------------------------------------------------------------------------
__global__ void k_sigma(const float* __restrict__ x,
                        const float* __restrict__ Ws,
                        const float* __restrict__ Wsb,
                        float* __restrict__ sig) {
    const int gid = blockIdx.x * 256 + threadIdx.x;  // (b*T+t)*H + h
    const int h  = gid & 7;
    const int bt = gid >> 3;
    const float* xr = x + (size_t)bt * DD;
    float acc = Wsb[h];
    for (int k = 0; k < DD; ++k) acc = fmaf(xr[k], Ws[k * HH + h], acc);
    const float sp = (acc > 20.f) ? acc : log1pf(__expf(acc));
    const int b = bt >> 10, t = bt & 1023;
    sig[((size_t)(b * HH + h) << 10) + t] = sp + 1e-6f;
}

// ---------------------------------------------------------------------------
// prior[b,h,t,s] = N(t-s; sigma) row-normalized
// ---------------------------------------------------------------------------
__global__ void k_prior(const float* __restrict__ sig,
                        float* __restrict__ prior) {
    __shared__ float buf[TT];
    __shared__ float red[8];
    const int row = blockIdx.x;           // (b*H + h)*T + t
    const int t   = row & 1023;
    const int tid = threadIdx.x;
    const float sg  = sig[row];
    const float inv = -0.5f / (sg * sg + 1e-6f);

    float s = 0.f;
    for (int i = tid; i < TT; i += 256) {
        const float d = (float)(t - i);
        const float e = __expf(d * d * inv);
        buf[i] = e; s += e;
    }
    s = wave_sum(s);
    if ((tid & 31) == 0) red[tid >> 5] = s;
    __syncthreads();
    float tot = red[0];
#pragma unroll
    for (int i = 1; i < 8; ++i) tot += red[i];
    const float invs = 1.0f / (tot + 1e-9f);
    float* p = prior + (size_t)row * TT;
    for (int i = tid; i < TT; i += 256) p[i] = buf[i] * invs;
}

// ---------------------------------------------------------------------------
// attn_out[b,t, h*64+dh] = sum_s series[b,h,t,s] * v[b,h,s,dh]
// Block = 8 waves = 8 t-tiles x full head width (DH=64).
// The V strip chunk (64 x 32 f16) is staged in LDS (double buffered) and
// shared by all 8 waves; the f32 A fragment is reused by 4 WMMAs per wave.
// ---------------------------------------------------------------------------
__global__ void k_attn_v(const float* __restrict__ series,
                         const _Float16* __restrict__ vt,
                         _Float16* __restrict__ ao) {
    __shared__ _Float16 lb[2][64 * 32];
    const int bh   = blockIdx.z;
    const int b    = bh >> 3, h = bh & 7;
    const int tid  = threadIdx.x;
    const int lane = tid & 31;
    const int w    = tid >> 5;
    const int m0 = (blockIdx.x << 7) + (w << 4);            // t-tile per wave
    const float*    srow = series + (size_t)bh * TT * TT;
    const _Float16* vb   = vt + (size_t)bh * DHH * TT;      // [DH][T]

    const int sr = tid >> 2, sc = (tid & 3) << 3;           // staging slot
    const _Float16* bsrc = vb + (size_t)sr * TT + sc;
    _Float16* bdst0 = &lb[0][sr * 32 + sc];
    _Float16* bdst1 = &lb[1][sr * 32 + sc];

    stage16(bsrc, bdst0);
    stage_wait();

    v8f acc[4] = {};
    for (int c = 0; c < (TT >> 5); ++c) {
        const int cur = c & 1;
        if (c + 1 < (TT >> 5)) stage16(bsrc + (c + 1) * 32, cur ? bdst0 : bdst1);
        const v16h a = load_a_f32(srow, TT, m0, c * 32, lane);
#pragma unroll
        for (int s = 0; s < 4; ++s) {
            const v16h bfr = load_b_f16t(&lb[cur][0], 32, 16 * s, 0, lane);
            acc[s] = wmma_f16(a, bfr, acc[s]);
        }
        stage_wait();
    }

    const int lo = (lane < 16) ? 0 : 8;
#pragma unroll
    for (int s = 0; s < 4; ++s) {
        const int n = 16 * s + (lane & 15);
#pragma unroll
        for (int r = 0; r < 8; ++r) {
            const int t = m0 + r + lo;
            ao[((size_t)(b * TT + t)) * DD + h * DHH + n] = (_Float16)acc[s][r];
        }
    }
}

// ---------------------------------------------------------------------------
extern "C" void kernel_launch(void* const* d_in, const int* in_sizes, int n_in,
                              void* d_out, int out_size, void* d_ws, size_t ws_size,
                              hipStream_t stream) {
    const float* x    = (const float*)d_in[0];
    const float* Wq_w = (const float*)d_in[1];
    const float* Wq_b = (const float*)d_in[2];
    const float* Wk_w = (const float*)d_in[3];
    const float* Wk_b = (const float*)d_in[4];
    const float* Wv_w = (const float*)d_in[5];
    const float* Wv_b = (const float*)d_in[6];
    const float* Ws_w = (const float*)d_in[7];
    const float* Ws_b = (const float*)d_in[8];
    const float* Wo_w = (const float*)d_in[9];
    const float* Wo_b = (const float*)d_in[10];

    float* out    = (float*)d_out;                 // [B,T,D]
    float* series = out + (size_t)BB * TT * DD;    // [B,H,T,T]
    float* prior  = series + (size_t)BB * HH * TT * TT;
    float* sig    = prior + (size_t)BB * HH * TT * TT;   // [B,H,T]

    char* ws = (char*)d_ws;
    _Float16* xh   = (_Float16*)ws;                         // 4 MB
    _Float16* wq_t = (_Float16*)(ws + (size_t)4 * 1048576); // 512 KB each
    _Float16* wk_t = wq_t + DD * DD;
    _Float16* wv_t = wk_t + DD * DD;
    _Float16* wo_t = wv_t + DD * DD;
    _Float16* qh   = (_Float16*)(ws + (size_t)6 * 1048576); // 4 MB each
    _Float16* kh   = qh + (size_t)BB * HH * TT * DHH;
    _Float16* vt   = kh + (size_t)BB * HH * TT * DHH;
    _Float16* ao   = vt + (size_t)BB * HH * TT * DHH;

    const int nX = BB * TT * DD;                 // 2097152
    k_f32_to_f16<<<(nX + 255) / 256, 256, 0, stream>>>(x, xh, nX);
    k_transpose_w<<<1024, 256, 0, stream>>>(Wq_w, wq_t);
    k_transpose_w<<<1024, 256, 0, stream>>>(Wk_w, wk_t);
    k_transpose_w<<<1024, 256, 0, stream>>>(Wv_w, wv_t);
    k_transpose_w<<<1024, 256, 0, stream>>>(Wo_w, wo_t);

    k_sigma<<<(BB * TT * HH) / 256, 256, 0, stream>>>(x, Ws_w, Ws_b, sig);

    // QKV projections: (4096/128) * (512/64) = 256 blocks
    k_gemm<<<256, 256, 0, stream>>>(xh, wq_t, Wq_b, qh, nullptr, 4096, 512, 512, 0);
    k_gemm<<<256, 256, 0, stream>>>(xh, wk_t, Wk_b, kh, nullptr, 4096, 512, 512, 0);
    k_gemm<<<256, 256, 0, stream>>>(xh, wv_t, Wv_b, vt, nullptr, 4096, 512, 512, 1);

    // QK^T: (64/8 m-groups) * 16 s-strips = 128 blocks per (b,h)
    k_qk_logits<<<dim3(128, 1, BB * HH), 256, 0, stream>>>(qh, kh, series);
    k_softmax<<<BB * HH * TT, 256, 0, stream>>>(series);
    k_prior<<<BB * HH * TT, 256, 0, stream>>>(sig, prior);
    // series @ V: 64/8 = 8 blocks per (b,h)
    k_attn_v<<<dim3(8, 1, BB * HH), 256, 0, stream>>>(series, vt, ao);

    // Output projection, f32 result straight into d_out
    k_gemm<<<256, 256, 0, stream>>>(ao, wo_t, Wo_b, nullptr, out, 4096, 512, 512, 3);
}
